// DynamicPTBlock_73280732004906
// MI455X (gfx1250) — compile-verified, matrix operand order checked by reading, not run on hardware
//
#include <hip/hip_runtime.h>
#include <hip/hip_bf16.h>

// ---------------- problem constants ----------------
#define BB 8
#define CC 32
#define PP 128
#define DD 256
#define HH 8
#define GG 512
#define RR 32          // DD/HH
#define NTOK (BB*CC*PP)   // 32768 rows of D

typedef _Float16 h16;
typedef __attribute__((ext_vector_type(16))) _Float16 v16h;
typedef __attribute__((ext_vector_type(8)))  _Float16 h8;
typedef __attribute__((ext_vector_type(8)))  float    v8f;

union frag_u { v16h v; h8 h[2]; };

__device__ __forceinline__ v8f wmma_f16(const frag_u& a, const frag_u& b, v8f c) {
  return __builtin_amdgcn_wmma_f32_16x16x32_f16(false, a.v, false, b.v, (short)0, c, false, false);
}

// Async 32B copy global->LDS (two b128 chunks). ASYNCcnt-tracked; the
// instruction offset is added to BOTH the LDS and global addresses.
__device__ __forceinline__ void async_copy_32B(const h16* gsrc, h16* lds_dst) {
  unsigned lds = (unsigned)(uintptr_t)lds_dst;   // low 32 bits of flat addr = LDS offset
  asm volatile(
      "global_load_async_to_lds_b128 %0, %1, off\n\t"
      "global_load_async_to_lds_b128 %0, %1, off offset:16"
      :: "v"(lds), "v"(gsrc) : "memory");
}

__device__ __forceinline__ void wait_async_all() {
  asm volatile("s_wait_asynccnt 0x0" ::: "memory");
}

// ---------------- elementwise / LN kernels ----------------

__global__ __launch_bounds__(256) void pack_f16_kernel(const float* __restrict__ in,
                                                       h16* __restrict__ out, int n) {
  int i = blockIdx.x * 256 + threadIdx.x;
  if (i < n) out[i] = (h16)in[i];
}

// One block per row of D=256. Writes f16 LayerNorm output.
__global__ __launch_bounds__(256) void ln_kernel(const float* __restrict__ x,
                                                 h16* __restrict__ out,
                                                 const float* __restrict__ g,
                                                 const float* __restrict__ b) {
  __shared__ float red[256];
  long row = blockIdx.x;
  int t = threadIdx.x;
  float v = x[row * DD + t];
  red[t] = v; __syncthreads();
  for (int s = 128; s > 0; s >>= 1) { if (t < s) red[t] += red[t + s]; __syncthreads(); }
  float mean = red[0] * (1.0f / DD);
  __syncthreads();
  float d = v - mean;
  red[t] = d * d; __syncthreads();
  for (int s = 128; s > 0; s >>= 1) { if (t < s) red[t] += red[t + s]; __syncthreads(); }
  float var = red[0] * (1.0f / DD);
  float y = d * rsqrtf(var + 1e-5f) * g[t] + b[t];
  out[row * DD + t] = (h16)y;
}

// L1-normalize relu'd topic scores over G=512; one block per row.
__global__ __launch_bounds__(256) void topicnorm_kernel(const float* __restrict__ qg32,
                                                        h16* __restrict__ qg16) {
  __shared__ float red[256];
  long row = blockIdx.x;
  int t = threadIdx.x;
  float a = qg32[row * GG + t];
  float c = qg32[row * GG + t + 256];
  red[t] = a + c; __syncthreads();
  for (int s = 128; s > 0; s >>= 1) { if (t < s) red[t] += red[t + s]; __syncthreads(); }
  float inv = 1.0f / fmaxf(red[0], 1e-6f);
  qg16[row * GG + t]       = (h16)(a * inv);
  qg16[row * GG + t + 256] = (h16)(c * inv);
}

// ---------------- generic batched WMMA GEMM ----------------
// C[m,n] = epi( alpha * sum_k A[m,k] * B[k,n] (+bias[n]) )
// A: f16 row-major (lda=K), staged via GLOBAL_LOAD_ASYNC_TO_LDS_B128.
// B: transB? (N,K) : (K,N); f16 or f32 source.
// Tile 64x64x32, 4 waves, each wave 32x32 (2x2 wmma frags).
// act: 0 none, 1 relu, 2 exact gelu.  mode: 0 store, 1 +=, 2 resid+, 3 sigmoid-gate blend.
__global__ __launch_bounds__(128)
void wmma_gemm_kernel(const h16* __restrict__ A,
                      const h16* __restrict__ B16, const float* __restrict__ B32,
                      int K, int N, int transB,
                      long strideA, long strideB, long strideO,
                      float alpha, const float* __restrict__ bias,
                      const float* __restrict__ resid, const float* __restrict__ aux,
                      float* __restrict__ out32, h16* __restrict__ out16,
                      int act, int mode) {
  __shared__ __align__(16) h16 As[64][40];
  __shared__ __align__(16) h16 Bs[64][40];   // stored (n_local, k_local)

  A += (long)blockIdx.z * strideA;
  if (B16) B16 += (long)blockIdx.z * strideB;
  if (B32) B32 += (long)blockIdx.z * strideB;
  long obase = (long)blockIdx.z * strideO;

  int n0 = blockIdx.x * 64, m0 = blockIdx.y * 64;
  int t = threadIdx.x, lane = t & 31, w = t >> 5;
  int wm = (w & 1) * 32, wn = (w >> 1) * 32;
  int half = lane >> 4, lm = lane & 15;

  v8f acc[2][2] = {};

  for (int k0 = 0; k0 < K; k0 += 32) {
    // --- stage A (64x32 f16) via async global->LDS DMA ---
    {
      int row = t >> 1, off = (t & 1) * 16;
      async_copy_32B(A + (long)(m0 + row) * K + k0 + off, &As[row][off]);
    }
    // --- stage B into (n,k) layout ---
    if (transB) {
      int row = t >> 1, off = (t & 1) * 16;     // row = n_local
      if (B32) {
        const float* src = B32 + (long)(n0 + row) * K + k0 + off;
        __builtin_prefetch(src + 32, 0, 0);     // next k-step (speculative)
        #pragma unroll
        for (int j = 0; j < 16; j++) Bs[row][off + j] = (h16)src[j];
      } else {
        async_copy_32B(B16 + (long)(n0 + row) * K + k0 + off, &Bs[row][off]);
      }
    } else {
      int kr = t >> 2, off = (t & 3) * 16;      // kr = k_local row, off = n offset
      if (B32) {
        const float* src = B32 + (long)(k0 + kr) * N + n0 + off;
        #pragma unroll
        for (int j = 0; j < 16; j++) Bs[off + j][kr] = (h16)src[j];
      } else {
        const h16* src = B16 + (long)(k0 + kr) * N + n0 + off;
        h8 v0 = *(const h8*)(src);
        h8 v1 = *(const h8*)(src + 8);
        #pragma unroll
        for (int j = 0; j < 8; j++) { Bs[off + j][kr] = v0[j]; Bs[off + 8 + j][kr] = v1[j]; }
      }
    }
    wait_async_all();        // ASYNCcnt -> 0 before signaling producers done
    __syncthreads();

    frag_u af[2], bf[2];
    #pragma unroll
    for (int i = 0; i < 2; i++) {
      int r = wm + i * 16 + lm, kk = half * 8;
      af[i].h[0] = *(const h8*)&As[r][kk];
      af[i].h[1] = *(const h8*)&As[r][kk + 16];
    }
    #pragma unroll
    for (int i = 0; i < 2; i++) {
      int r = wn + i * 16 + lm, kk = half * 8;
      bf[i].h[0] = *(const h8*)&Bs[r][kk];
      bf[i].h[1] = *(const h8*)&Bs[r][kk + 16];
    }
    #pragma unroll
    for (int i = 0; i < 2; i++)
      #pragma unroll
      for (int j = 0; j < 2; j++)
        acc[i][j] = wmma_f16(af[i], bf[j], acc[i][j]);
    __syncthreads();
  }

  // --- epilogue ---
  #pragma unroll
  for (int i = 0; i < 2; i++) {
    #pragma unroll
    for (int j = 0; j < 2; j++) {
      int gn = n0 + wn + j * 16 + lm;
      int mb = m0 + wm + i * 16 + half * 8;
      #pragma unroll
      for (int g = 0; g < 8; g++) {
        long idx = obase + (long)(mb + g) * N + gn;
        float v = acc[i][j][g] * alpha;
        if (bias) v += bias[gn];
        if (act == 1)      v = v > 0.0f ? v : 0.0f;
        else if (act == 2) v = 0.5f * v * (1.0f + erff(v * 0.70710678118654752f));
        if (mode == 3) {
          float s = 1.0f / (1.0f + __expf(-v));
          out32[idx] = s * resid[idx] + (1.0f - s) * aux[idx];
        } else if (mode == 2) {
          out32[idx] = resid[idx] + v;
        } else if (mode == 1) {
          out32[idx] += v;
        } else {
          if (out32) out32[idx] = v;
          if (out16) out16[idx] = (h16)v;
        }
      }
    }
  }
}

// ---------------- time attention: Asum[b,c] = (1/H) * sum_h softmax(qu_h @ qv_h^T) ----------------
// One block per (b,c); 8 waves, wave w owns row strip [w*16, w*16+16) of P.
// Scale 1/sqrt(R) is pre-folded into qu. Fragments load straight from global (K-contiguous).
__global__ __launch_bounds__(256)
void time_attn_kernel(const h16* __restrict__ qu, const h16* __restrict__ qv,
                      h16* __restrict__ asum) {
  int bc = blockIdx.x;
  int t = threadIdx.x, lane = t & 31, w = t >> 5;
  int half = lane >> 4, lm = lane & 15;
  int m0 = w * 16;
  const h16* qub = qu + (long)bc * PP * DD;
  const h16* qvb = qv + (long)bc * PP * DD;

  v8f accsum[8] = {};
  for (int h = 0; h < HH; h++) {
    frag_u a;
    const h16* ap = qub + (long)(m0 + lm) * DD + h * RR + half * 8;
    a.h[0] = *(const h8*)ap; a.h[1] = *(const h8*)(ap + 16);
    v8f s[8];
    #pragma unroll
    for (int tt = 0; tt < 8; tt++) {
      frag_u b;
      const h16* bp = qvb + (long)(tt * 16 + lm) * DD + h * RR + half * 8;
      b.h[0] = *(const h8*)bp; b.h[1] = *(const h8*)(bp + 16);
      v8f z = {};
      s[tt] = wmma_f16(a, b, z);
    }
    // row softmax: row = m0 + half*8 + g, 128 cols spread over 8 tiles x 16 lanes
    #pragma unroll
    for (int g = 0; g < 8; g++) {
      float mx = -3.0e38f;
      #pragma unroll
      for (int tt = 0; tt < 8; tt++) mx = fmaxf(mx, s[tt][g]);
      mx = fmaxf(mx, __shfl_xor(mx, 1, 32));
      mx = fmaxf(mx, __shfl_xor(mx, 2, 32));
      mx = fmaxf(mx, __shfl_xor(mx, 4, 32));
      mx = fmaxf(mx, __shfl_xor(mx, 8, 32));
      float e[8], sum = 0.0f;
      #pragma unroll
      for (int tt = 0; tt < 8; tt++) { e[tt] = __expf(s[tt][g] - mx); sum += e[tt]; }
      sum += __shfl_xor(sum, 1, 32);
      sum += __shfl_xor(sum, 2, 32);
      sum += __shfl_xor(sum, 4, 32);
      sum += __shfl_xor(sum, 8, 32);
      float inv = 1.0f / sum;
      #pragma unroll
      for (int tt = 0; tt < 8; tt++) accsum[tt][g] += e[tt] * inv;
    }
  }
  const float ih = 1.0f / HH;
  #pragma unroll
  for (int tt = 0; tt < 8; tt++)
    #pragma unroll
    for (int g = 0; g < 8; g++) {
      int row = m0 + half * 8 + g, col = tt * 16 + lm;
      asum[(long)bc * PP * PP + (long)row * PP + col] = (h16)(accsum[tt][g] * ih);
    }
}

// ---------------- channel attention: msg += (1/H) sum_h softmax(cu_h cv_h^T) @ Qp ----------------
// One block per (b,p); wave == head. 32x32 scores; head-sum via LDS atomics; second GEMM in-block.
__global__ __launch_bounds__(256)
void chan_attn_kernel(const h16* __restrict__ cu, const h16* __restrict__ cv,
                      const h16* __restrict__ qzn, float* __restrict__ msg) {
  __shared__ float Acs[32][33];
  int bp = blockIdx.x, b = bp >> 7, p = bp & (PP - 1);
  int t = threadIdx.x, lane = t & 31, w = t >> 5;   // w = head
  int half = lane >> 4, lm = lane & 15;

  for (int i = t; i < 32 * 33; i += 256) ((float*)Acs)[i] = 0.0f;
  __syncthreads();

  frag_u a[2], bf[2];
  #pragma unroll
  for (int i = 0; i < 2; i++) {
    const h16* ap = cu + ((long)(b * CC + i * 16 + lm) * PP + p) * DD + w * RR + half * 8;
    a[i].h[0] = *(const h8*)ap; a[i].h[1] = *(const h8*)(ap + 16);
    const h16* bp2 = cv + ((long)(b * CC + i * 16 + lm) * PP + p) * DD + w * RR + half * 8;
    bf[i].h[0] = *(const h8*)bp2; bf[i].h[1] = *(const h8*)(bp2 + 16);
  }
  v8f s[2][2];
  #pragma unroll
  for (int i = 0; i < 2; i++)
    #pragma unroll
    for (int j = 0; j < 2; j++) { v8f z = {}; s[i][j] = wmma_f16(a[i], bf[j], z); }

  // softmax rows (32 cols), scale pre-folded into cu; add (1/H)*softmax into shared Acs
  #pragma unroll
  for (int i = 0; i < 2; i++)
    #pragma unroll
    for (int g = 0; g < 8; g++) {
      float mx = fmaxf(s[i][0][g], s[i][1][g]);
      mx = fmaxf(mx, __shfl_xor(mx, 1, 32));
      mx = fmaxf(mx, __shfl_xor(mx, 2, 32));
      mx = fmaxf(mx, __shfl_xor(mx, 4, 32));
      mx = fmaxf(mx, __shfl_xor(mx, 8, 32));
      float e0 = __expf(s[i][0][g] - mx), e1 = __expf(s[i][1][g] - mx);
      float sum = e0 + e1;
      sum += __shfl_xor(sum, 1, 32);
      sum += __shfl_xor(sum, 2, 32);
      sum += __shfl_xor(sum, 4, 32);
      sum += __shfl_xor(sum, 8, 32);
      float sc = 1.0f / (sum * (float)HH);
      int r = i * 16 + half * 8 + g;
      atomicAdd(&Acs[r][lm],      e0 * sc);
      atomicAdd(&Acs[r][16 + lm], e1 * sc);
    }
  __syncthreads();

  // second GEMM: (32x32) @ Qp(32x256); wave w covers cols [w*32, w*32+32)
  frag_u aa[2];
  #pragma unroll
  for (int i = 0; i < 2; i++) {
    int r = i * 16 + lm, kk = half * 8;
    #pragma unroll
    for (int j = 0; j < 8; j++) aa[i].v[j] = (h16)Acs[r][kk + j];
    #pragma unroll
    for (int j = 0; j < 8; j++) aa[i].v[8 + j] = (h16)Acs[r][16 + kk + j];
  }
  v8f o[2][2] = {};
  #pragma unroll
  for (int j = 0; j < 2; j++) {
    frag_u bb;
    int d = w * 32 + j * 16 + lm;
    #pragma unroll
    for (int q = 0; q < 8; q++) {
      int f = half * 8 + q;
      bb.v[q] = qzn[((long)(b * CC + f) * PP + p) * DD + d];
    }
    #pragma unroll
    for (int q = 0; q < 8; q++) {
      int f = 16 + half * 8 + q;
      bb.v[8 + q] = qzn[((long)(b * CC + f) * PP + p) * DD + d];
    }
    #pragma unroll
    for (int i = 0; i < 2; i++) o[i][j] = wmma_f16(aa[i], bb, o[i][j]);
  }
  #pragma unroll
  for (int i = 0; i < 2; i++)
    #pragma unroll
    for (int j = 0; j < 2; j++)
      #pragma unroll
      for (int g = 0; g < 8; g++) {
        int c = i * 16 + half * 8 + g;
        int d = w * 32 + j * 16 + lm;
        long idx = ((long)(b * CC + c) * PP + p) * DD + d;
        msg[idx] += o[i][j][g];
      }
}

// ---------------- host orchestration ----------------

extern "C" void kernel_launch(void* const* d_in, const int* in_sizes, int n_in,
                              void* d_out, int out_size, void* d_ws, size_t ws_size,
                              hipStream_t stream) {
  (void)in_sizes; (void)n_in; (void)out_size; (void)ws_size;
  const float* qz     = (const float*)d_in[0];
  const float* unary  = (const float*)d_in[1];
  const float* time_u = (const float*)d_in[2];
  const float* time_v = (const float*)d_in[3];
  const float* chan_u = (const float*)d_in[4];
  const float* chan_v = (const float*)d_in[5];
  const float* topic  = (const float*)d_in[6];
  const float* ln_g   = (const float*)d_in[7];
  const float* ln_b   = (const float*)d_in[8];
  const float* gate_w = (const float*)d_in[9];
  const float* gate_b = (const float*)d_in[10];
  const float* out_w1 = (const float*)d_in[11];
  const float* out_b1 = (const float*)d_in[12];
  const float* out_w2 = (const float*)d_in[13];
  const float* out_b2 = (const float*)d_in[14];
  float* out = (float*)d_out;

  size_t off = 0;
  char* base = (char*)d_ws;
  auto alloc = [&](size_t bytes) { void* p = base + off; off += (bytes + 255) & ~(size_t)255; return p; };
  h16*   qzn16  = (h16*)  alloc((size_t)NTOK * DD * 2);
  h16*   qz16   = (h16*)  alloc((size_t)NTOK * DD * 2);
  h16*   qu16   = (h16*)  alloc((size_t)NTOK * DD * 2);
  h16*   qv16   = (h16*)  alloc((size_t)NTOK * DD * 2);
  h16*   cu16   = (h16*)  alloc((size_t)NTOK * DD * 2);
  h16*   cv16   = (h16*)  alloc((size_t)NTOK * DD * 2);
  h16*   asum16 = (h16*)  alloc((size_t)BB * CC * PP * PP * 2);
  float* qg32   = (float*)alloc((size_t)NTOK * GG * 4);
  h16*   qg16   = (h16*)  alloc((size_t)NTOK * GG * 2);
  float* msg    = (float*)alloc((size_t)NTOK * DD * 4);
  float* qz2    = (float*)alloc((size_t)NTOK * DD * 4);
  h16*   hn16   = (h16*)  alloc((size_t)NTOK * DD * 2);
  h16*   h116   = (h16*)  alloc((size_t)NTOK * DD * 2);

  const float scale = 0.17677669529663687f;  // 1/sqrt(R)
  const long sTok = (long)CC * PP * DD;      // per-batch A/out stride (tokens x D)
  const long sW   = (long)DD * DD;           // per-batch DxD weight stride

  // 1) LayerNorm(qz) -> f16 ; raw qz -> f16
  ln_kernel<<<NTOK, 256, 0, stream>>>(qz, qzn16, ln_g, ln_b);
  pack_f16_kernel<<<NTOK, 256, 0, stream>>>(qz, qz16, NTOK * DD);

  // 2) projections (batched over B): qu/qv/cu/cv = qzn @ W^T  (scale folded into qu,cu)
  dim3 gproj(DD / 64, (CC * PP) / 64, BB);
  wmma_gemm_kernel<<<gproj, 128, 0, stream>>>(qzn16, nullptr, time_u, DD, DD, 1,
      sTok, sW, sTok, scale, nullptr, nullptr, nullptr, nullptr, qu16, 0, 0);
  wmma_gemm_kernel<<<gproj, 128, 0, stream>>>(qzn16, nullptr, time_v, DD, DD, 1,
      sTok, sW, sTok, 1.0f, nullptr, nullptr, nullptr, nullptr, qv16, 0, 0);
  wmma_gemm_kernel<<<gproj, 128, 0, stream>>>(qzn16, nullptr, chan_u, DD, DD, 1,
      sTok, sW, sTok, scale, nullptr, nullptr, nullptr, nullptr, cu16, 0, 0);
  wmma_gemm_kernel<<<gproj, 128, 0, stream>>>(qzn16, nullptr, chan_v, DD, DD, 1,
      sTok, sW, sTok, 1.0f, nullptr, nullptr, nullptr, nullptr, cv16, 0, 0);

  // 3) time attention weights, then msg = unary + Asum @ qzn  (batched over b,c)
  time_attn_kernel<<<BB * CC, 256, 0, stream>>>(qu16, qv16, asum16);
  dim3 gt2(DD / 64, PP / 64, BB * CC);
  wmma_gemm_kernel<<<gt2, 128, 0, stream>>>(asum16, qzn16, nullptr, PP, DD, 0,
      (long)PP * PP, (long)PP * DD, (long)PP * DD, 1.0f, nullptr, unary, nullptr, msg, nullptr, 0, 2);

  // 4) channel attention: msg += m_c
  chan_attn_kernel<<<BB * PP, 256, 0, stream>>>(cu16, cv16, qzn16, msg);

  // 5) topic: qg = relu(qzn @ topic^T); L1-normalize; msg += qg @ topic
  dim3 gqg(GG / 64, (CC * PP) / 64, BB);
  wmma_gemm_kernel<<<gqg, 128, 0, stream>>>(qzn16, nullptr, topic, DD, GG, 1,
      sTok, (long)GG * DD, (long)CC * PP * GG, 1.0f, nullptr, nullptr, nullptr, qg32, nullptr, 1, 0);
  topicnorm_kernel<<<NTOK, 256, 0, stream>>>(qg32, qg16);
  dim3 gmg(DD / 64, (CC * PP) / 64, BB);
  wmma_gemm_kernel<<<gmg, 128, 0, stream>>>(qg16, nullptr, topic, GG, DD, 0,
      (long)CC * PP * GG, (long)GG * DD, sTok, 1.0f, nullptr, nullptr, nullptr, msg, nullptr, 0, 1);

  // 6) gate: qz2 = sigmoid(qz @ gate_w^T + b) * qz + (1-s) * msg   (single big GEMM)
  dim3 gbig(DD / 64, NTOK / 64, 1);
  wmma_gemm_kernel<<<gbig, 128, 0, stream>>>(qz16, nullptr, gate_w, DD, DD, 1,
      0, 0, 0, 1.0f, gate_b, qz, msg, qz2, nullptr, 0, 3);

  // 7) out = qz2 + W2(gelu(W1 LN(qz2) + b1)) + b2
  ln_kernel<<<NTOK, 256, 0, stream>>>(qz2, hn16, ln_g, ln_b);
  wmma_gemm_kernel<<<gbig, 128, 0, stream>>>(hn16, nullptr, out_w1, DD, DD, 1,
      0, 0, 0, 1.0f, out_b1, nullptr, nullptr, nullptr, h116, 2, 0);
  wmma_gemm_kernel<<<gbig, 128, 0, stream>>>(h116, nullptr, out_w2, DD, DD, 1,
      0, 0, 0, 1.0f, out_b2, qz2, nullptr, out, nullptr, 0, 2);
}
